// AdditiveAttention_34789235098190
// MI455X (gfx1250) — matrix operand order repaired and checked by reading, not compiled, not run on hardware
//
#include <hip/hip_runtime.h>
#include <hip/hip_bf16.h>
#include <cstdint>

// Problem constants (from reference): B=64, S=2048, D=512, U=512
#define BB 64
#define SS 2048
#define DD 512
#define UU 512
#define MTILE 64          // S-rows per workgroup in the score kernel
#define NTHREADS 256      // 8 waves (wave32)
#define APAD 520          // LDS row stride in bf16 elements (512 + 8 pad)

typedef __bf16 v16bf __attribute__((ext_vector_type(16)));
typedef float  v8f   __attribute__((ext_vector_type(8)));

__device__ __forceinline__ uint16_t f32_to_bf16(float f) {
    uint32_t u = __float_as_uint(f);
    uint32_t r = u + 0x7FFFu + ((u >> 16) & 1u);   // round-to-nearest-even
    return (uint16_t)(r >> 16);
}

// Branch-free fast tanh: tanh(x) = 1 - 2/(exp(2x)+1). Clamp keeps exp finite;
// tanh saturates to +/-1 well before |x|=9.  Uses v_exp_f32 + v_rcp_f32.
__device__ __forceinline__ float fast_tanh(float x) {
    float xc = fminf(fmaxf(x, -9.0f), 9.0f);
    float e  = __expf(2.0f * xc);
    return fmaf(-2.0f, __builtin_amdgcn_rcpf(e + 1.0f), 1.0f);
}

// CDNA5 async global->LDS copy (ASYNCcnt-tracked), per-lane 16 bytes.
__device__ __forceinline__ void async_copy_b128(uint32_t lds_addr, const void* gaddr) {
    asm volatile("global_load_async_to_lds_b128 %0, %1, off"
                 :: "v"(lds_addr), "v"((uint64_t)(uintptr_t)gaddr)
                 : "memory");
}
__device__ __forceinline__ void wait_async0() {
    asm volatile("s_wait_asynccnt 0x0" ::: "memory");
}

// ---------------------------------------------------------------------------
// Pack W2 (DxU f32, row major) into bf16 WMMA B-fragments.
// Fragment f = ntile*16 + kt covers B-tile K=[kt*32,+32), N=[ntile*16,+16).
// Contiguous-K order (32x16 16-bit B, same order as A): lane l element e holds
//   K = kt*32 + (l<16 ? 0 : 16) + e,  N = ntile*16 + (l & 15)
// Stored as w2p[f*512 + lane*16 + e]; the 16 fragments of one ntile are a
// contiguous 16 KB block (ideal for async bulk copy to LDS).
// ---------------------------------------------------------------------------
__global__ __launch_bounds__(256) void pack_w2_kernel(const float* __restrict__ W2,
                                                      uint16_t* __restrict__ w2p) {
    int gid = blockIdx.x * 256 + threadIdx.x;      // 0 .. 512*512-1
    int f    = gid >> 9;
    int r    = gid & 511;
    int lane = r >> 4;
    int e    = r & 15;
    int ntile = f >> 4;
    int kt    = f & 15;
    int k = kt * 32 + ((lane < 16) ? 0 : 16) + e;
    int n = ntile * 16 + (lane & 15);
    w2p[gid] = f32_to_bf16(W2[k * UU + n]);
}

// ---------------------------------------------------------------------------
// Q = query @ W1   (64x512 @ 512x512) -- tiny, scalar FMA is fine.
// ---------------------------------------------------------------------------
__global__ __launch_bounds__(256) void compute_q_kernel(const float* __restrict__ query,
                                                        const float* __restrict__ W1,
                                                        float* __restrict__ Qf) {
    int gid = blockIdx.x * 256 + threadIdx.x;      // 0 .. 64*512-1
    int b = gid >> 9;
    int u = gid & 511;
    const float* q = query + b * DD;
    float acc = 0.f;
    for (int d = 0; d < DD; ++d)
        acc = fmaf(q[d], W1[d * UU + u], acc);
    Qf[gid] = acc;
}

// ---------------------------------------------------------------------------
// Fused score kernel: per (batch, s-tile of 64 rows)
//   V = values_tile @ W2  (bf16 WMMA, f32 accum, K=512)
//   score[row] = sum_n tanh(Q[b,n] + V[row,n]) * V1[n]
// 8 waves: wave -> (m_tile = wave>>1, n_half = wave&1). Each wave covers 16
// n-tiles of 16 columns, 16 K-steps each => 256 WMMAs/wave.
// B fragments are double-buffered in LDS via global_load_async_to_lds_b128:
// the 4 waves of one n-half cooperatively copy the next ntile's 16 KB block
// while the current one feeds the WMMA pipe.
// ---------------------------------------------------------------------------
__global__ __launch_bounds__(NTHREADS) void score_kernel(
        const float* __restrict__ values,
        const uint16_t* __restrict__ w2p,
        const float* __restrict__ Qf,
        const float* __restrict__ V1,
        float* __restrict__ scores) {
    __shared__ alignas(16) uint16_t Atile[MTILE * APAD];          // ~65 KB
    __shared__ alignas(16) uint16_t Btile[2][2][16 * 512];        //  64 KB
    __shared__ float sScore[MTILE];

    const int tid = threadIdx.x;
    const int b   = blockIdx.y;
    const int s0  = blockIdx.x * MTILE;

    if (tid < MTILE) sScore[tid] = 0.f;

    const int wave  = tid >> 5;
    const int lane  = tid & 31;
    const int mt    = wave >> 1;          // 0..3 : 16-row M tile / B-copy quarter
    const int nhalf = wave & 1;           // 0..1 : which half of the 32 n-tiles
    const int lmod  = lane & 15;
    const int lhi   = lane >> 4;          // 0/1 : lane half

    // Issue async copy of B-fragment block for a given nti into buffer `buf`.
    const uint32_t bLds = (uint32_t)(size_t)(&Btile[0][0][0]);
    auto issueB = [&](int nti, int buf) {
        const int ntile = nhalf * 16 + nti;
        const char* gsrc = (const char*)(w2p + (size_t)ntile * 16 * 512);
        const uint32_t base = bLds + (uint32_t)(((buf * 2 + nhalf) * 16 * 512) * 2);
        #pragma unroll
        for (int i = 0; i < 8; ++i) {
            uint32_t off = (uint32_t)(mt * 4096 + i * 512 + lane * 16);
            async_copy_b128(base + off, gsrc + off);
        }
    };

    // Stage values tile (MTILE x DD f32) -> LDS bf16, padded rows.
    const float4* src = reinterpret_cast<const float4*>(
        values + (size_t)b * SS * DD + (size_t)s0 * DD);
    #pragma unroll
    for (int i = 0; i < (MTILE * DD / 4) / NTHREADS; ++i) {
        int idx4 = tid + i * NTHREADS;   // float4 index within tile
        int row  = idx4 >> 7;            // DD/4 = 128 float4 per row
        int c4   = idx4 & 127;
        float4 v = src[idx4];
        uint16_t* dst = &Atile[row * APAD + c4 * 4];
        dst[0] = f32_to_bf16(v.x);
        dst[1] = f32_to_bf16(v.y);
        dst[2] = f32_to_bf16(v.z);
        dst[3] = f32_to_bf16(v.w);
    }

    // Prefetch first B block while A staging drains.
    issueB(0, 0);

    // A fragment base for this lane: row = mt*16 + lmod, K sub-offset lhi*16
    const uint16_t* aBase = &Atile[(mt * 16 + lmod) * APAD + lhi * 16];

    float scorep[8];
    #pragma unroll
    for (int v = 0; v < 8; ++v) scorep[v] = 0.f;

    for (int nti = 0; nti < 16; ++nti) {
        const int cur = nti & 1;
        // My async copies (for buffer `cur`) done, then WG-wide visibility.
        wait_async0();
        __syncthreads();
        // Overlap: start fetching next ntile's B block into the other buffer.
        // Safe: buffer cur^1 was last read in iteration nti-1, and every wave
        // passed the barrier above only after finishing that compute.
        if (nti < 15) issueB(nti + 1, cur ^ 1);

        const int ntile = nhalf * 16 + nti;
        const int col   = ntile * 16 + lmod;      // N column held by this lane
        const float qv  = Qf[b * UU + col];
        const float v1c = V1[col];

        v8f acc = {0.f, 0.f, 0.f, 0.f, 0.f, 0.f, 0.f, 0.f};
        #pragma unroll 4
        for (int kt = 0; kt < 16; ++kt) {
            union { uint4 u4[2]; v16bf v; } A, Bm;
            // A: elems 0..15 -> K = kt*32+lhi*16+0..15
            const uint4* ap = reinterpret_cast<const uint4*>(aBase + kt * 32);
            A.u4[0] = ap[0];
            A.u4[1] = ap[1];
            // B: fragment from the async-staged LDS buffer.
            const uint4* bp = reinterpret_cast<const uint4*>(
                &Btile[cur][nhalf][kt * 512 + lane * 16]);
            Bm.u4[0] = bp[0];
            Bm.u4[1] = bp[1];
            acc = __builtin_amdgcn_wmma_f32_16x16x32_bf16(
                false, A.v, false, Bm.v, (short)0, acc, false, false);
        }
        // Fuse tanh(Q+V)*V1 on the accumulator fragment.
        // C/D layout: lane holds N=lmod; VGPR v holds row (lhi ? v+8 : v).
        #pragma unroll
        for (int v = 0; v < 8; ++v)
            scorep[v] += fast_tanh(qv + acc[v]) * v1c;
    }

    // Reduce partial row scores (16 lanes per row half) into LDS.
    #pragma unroll
    for (int v = 0; v < 8; ++v) {
        int rowl = lhi ? (v + 8) : v;
        atomicAdd(&sScore[mt * 16 + rowl], scorep[v]);
    }
    __syncthreads();
    if (tid < MTILE)
        scores[(size_t)b * SS + s0 + tid] = sScore[tid];
}

// ---------------------------------------------------------------------------
// Softmax over S per batch, in place. One WG of 256 threads per batch.
// ---------------------------------------------------------------------------
__global__ __launch_bounds__(256) void softmax_kernel(float* __restrict__ scores) {
    __shared__ float red[256];
    const int b = blockIdx.x;
    const int tid = threadIdx.x;
    float* s = scores + (size_t)b * SS;

    float loc[SS / 256];
    float m = -3.0e38f;
    #pragma unroll
    for (int i = 0; i < SS / 256; ++i) {
        loc[i] = s[tid + i * 256];
        m = fmaxf(m, loc[i]);
    }
    red[tid] = m; __syncthreads();
    for (int o = 128; o > 0; o >>= 1) {
        if (tid < o) red[tid] = fmaxf(red[tid], red[tid + o]);
        __syncthreads();
    }
    m = red[0]; __syncthreads();

    float sum = 0.f;
    #pragma unroll
    for (int i = 0; i < SS / 256; ++i) {
        loc[i] = __expf(loc[i] - m);
        sum += loc[i];
    }
    red[tid] = sum; __syncthreads();
    for (int o = 128; o > 0; o >>= 1) {
        if (tid < o) red[tid] += red[tid + o];
        __syncthreads();
    }
    const float inv = 1.f / red[0];
    #pragma unroll
    for (int i = 0; i < SS / 256; ++i)
        s[tid + i * 256] = loc[i] * inv;
}

// ---------------------------------------------------------------------------
// context[b,d] = sum_s attn[b,s] * values[b,s,d]. One WG per batch.
// attn staged in LDS; values reads fully coalesced.
// ---------------------------------------------------------------------------
__global__ __launch_bounds__(256) void context_kernel(const float* __restrict__ values,
                                                      const float* __restrict__ attn,
                                                      float* __restrict__ out) {
    __shared__ float aLds[SS];
    const int b = blockIdx.x;
    const int tid = threadIdx.x;
    for (int i = tid; i < SS; i += 256)
        aLds[i] = attn[(size_t)b * SS + i];
    __syncthreads();

    const float* vb = values + (size_t)b * SS * DD;
    float acc0 = 0.f, acc1 = 0.f;
    const int d0 = tid, d1 = tid + 256;
    for (int s = 0; s < SS; ++s) {
        const float a = aLds[s];
        acc0 = fmaf(a, vb[(size_t)s * DD + d0], acc0);
        acc1 = fmaf(a, vb[(size_t)s * DD + d1], acc1);
    }
    out[(size_t)b * DD + d0] = acc0;
    out[(size_t)b * DD + d1] = acc1;
}

// ---------------------------------------------------------------------------
extern "C" void kernel_launch(void* const* d_in, const int* in_sizes, int n_in,
                              void* d_out, int out_size, void* d_ws, size_t ws_size,
                              hipStream_t stream) {
    const float* query  = (const float*)d_in[0];   // (B, D)
    const float* values = (const float*)d_in[1];   // (B, S, D)
    const float* W1     = (const float*)d_in[2];   // (D, U)
    const float* W2     = (const float*)d_in[3];   // (D, U)
    const float* V1     = (const float*)d_in[4];   // (U, 1)
    float* out = (float*)d_out;                    // (B, D)

    char* ws = (char*)d_ws;
    uint16_t* w2p    = (uint16_t*)ws;                          // 512*512*2 = 512 KB
    float*    Qf     = (float*)(ws + 524288);                  // 64*512*4  = 128 KB
    float*    scores = (float*)(ws + 524288 + 131072);         // 64*2048*4 = 512 KB

    pack_w2_kernel  <<<dim3((DD * UU) / 256), dim3(256), 0, stream>>>(W2, w2p);
    compute_q_kernel<<<dim3((BB * UU) / 256), dim3(256), 0, stream>>>(query, W1, Qf);
    score_kernel    <<<dim3(SS / MTILE, BB), dim3(NTHREADS), 0, stream>>>(values, w2p, Qf, V1, scores);
    softmax_kernel  <<<dim3(BB), dim3(256), 0, stream>>>(scores);
    context_kernel  <<<dim3(BB), dim3(256), 0, stream>>>(values, scores, out);
}
